// vMFLoss_81707457839465
// MI455X (gfx1250) — compile-verified, matrix operand order checked by reading, not run on hardware
//
#include <hip/hip_runtime.h>
#include <hip/hip_fp16.h>
#include <math.h>

typedef __attribute__((ext_vector_type(16))) _Float16 v16h;
typedef __attribute__((ext_vector_type(8)))  _Float16 v8h;
typedef __attribute__((ext_vector_type(8)))  float    v8f;

#define B_    8
#define K_    19
#define KP    32          // padded class dim for WMMA
#define P_    16384
#define C_    256
#define N_    (B_*P_)
#define KA_   10.0f
#define ITERS_ 10
#define CETH  0.22314355f // -log(0.8)

#define NT_PER_B   (P_/16)                  // 1024 n-tiles (16 wide) per batch
#define BLKSTRIDE  ((size_t)NT_PER_B * 512) // f16 elems per (b,cc) slab of featB

// ---------------- WMMA fragment loaders (wave32, 16x16x32 f16) ----------------
// A matrix 16(M) x 32(Kdim), row-major source with leading dim ld:
// lane<16: row M=lane, holds K=0..7 (elems 0..7) and K=16..23 (elems 8..15)
// lane>=16: row M=lane-16, holds K=8..15 and K=24..31
__device__ inline v16h frag_a_rm(const _Float16* __restrict__ src, int ld) {
  const int lane = threadIdx.x & 31;
  const _Float16* r = src + (lane & 15) * (size_t)ld + ((lane >> 4) << 3);
  v16h a;
#pragma unroll
  for (int i = 0; i < 8; ++i) { a[i] = r[i]; a[8 + i] = r[16 + i]; }
  return a;
}
// B matrix 32(Kdim) x 16(N), row-major source [k][n] with leading dim ld (gathered)
__device__ inline v16h frag_b_rm(const _Float16* __restrict__ src, int ld) {
  const int lane = threadIdx.x & 31;
  const int col = lane & 15;
  const int k0 = (lane >> 4) << 3;
  v16h b;
#pragma unroll
  for (int i = 0; i < 8; ++i) {
    b[i]     = src[(size_t)(k0 + i) * ld + col];
    b[8 + i] = src[(size_t)(k0 + 16 + i) * ld + col];
  }
  return b;
}
// B fragment from pre-swizzled block (512 f16): one contiguous 32B load per lane
__device__ inline v16h frag_b_sw(const _Float16* __restrict__ block) {
  const int lane = threadIdx.x & 31;
  return *reinterpret_cast<const v16h*>(block + lane * 16);
}

// ---------------- K1: segment-sum prototypes + L2 norm + f16 convert ----------
// grid: B_ * (P_/1024) blocks, 256 threads; 4 positions / thread
__global__ void k_proto_norm(const float* __restrict__ feat, const int* __restrict__ target,
                             float* __restrict__ ws_proto, float* __restrict__ ws_counts,
                             _Float16* __restrict__ featA) {
  __shared__ float proto_l[20 * C_];
  __shared__ float counts_l[20];
  const int b   = blockIdx.x / (P_ / 1024);
  const int p0  = (blockIdx.x % (P_ / 1024)) * 1024;
  const int tid = threadIdx.x;
  for (int i = tid; i < 20 * C_; i += 256) proto_l[i] = 0.f;
  if (tid < 20) counts_l[tid] = 0.f;
  __syncthreads();

  int p[4], cls[4]; float sumsq[4];
#pragma unroll
  for (int j = 0; j < 4; ++j) {
    p[j] = p0 + j * 256 + tid;
    int t = target[b * P_ + p[j]];
    cls[j] = ((unsigned)t < (unsigned)K_) ? t : K_;
    sumsq[j] = 0.f;
    atomicAdd(&counts_l[cls[j]], 1.f);
  }
  for (int c = 0; c < C_; ++c) {
    const float* row = feat + ((size_t)(b * C_ + c)) * P_;
#pragma unroll
    for (int j = 0; j < 4; ++j) {
      float f = row[p[j]];
      sumsq[j] += f * f;
      atomicAdd(&proto_l[cls[j] * C_ + c], f);  // LDS ds_add_f32
    }
  }
  __syncthreads();
  for (int i = tid; i < 20 * C_; i += 256) atomicAdd(&ws_proto[i], proto_l[i]);
  if (tid < 20) atomicAdd(&ws_counts[tid], counts_l[tid]);

  float inv[4];
#pragma unroll
  for (int j = 0; j < 4; ++j) inv[j] = 1.f / fmaxf(sqrtf(sumsq[j]), 1e-12f);
  for (int c = 0; c < C_; ++c) {
    const float*  row  = feat  + ((size_t)(b * C_ + c)) * P_;
    _Float16*     orow = featA + ((size_t)(b * C_ + c)) * P_;
#pragma unroll
    for (int j = 0; j < 4; ++j) orow[p[j]] = (_Float16)(row[p[j]] * inv[j]);
  }
}

// ---------------- K1b: swizzle featA -> WMMA-B-fragment layout (featB) --------
// one thread per 8 consecutive output f16 (one 16B store); reads are L2-hot.
// featB block (b, cc, n_tile) = 512 f16 = [lane 0..31][elem 0..15]
__global__ void k_swizzle(const _Float16* __restrict__ featA, _Float16* __restrict__ featB) {
  const size_t gid = (size_t)blockIdx.x * 256 + threadIdx.x;  // B*C*P/8 total
  const size_t blk = gid >> 6;           // 512-elem block id
  const int w  = (int)(gid & 63);        // 8-elem group within block
  const int l  = w >> 1;                 // lane 0..31
  const int e0 = (w & 1) * 8;            // elem offset 0 or 8
  const int ni = l & 15;
  const int k0 = (l >> 4) << 3;
  const int cbase = (e0 == 0) ? k0 : k0 + 16;   // 8 consecutive channels
  const int b       = (int)(blk / (8 * (size_t)NT_PER_B));
  const size_t rem  = blk % (8 * (size_t)NT_PER_B);
  const int cc      = (int)(rem / NT_PER_B);
  const int n_tile  = (int)(rem % NT_PER_B);
  const int n       = n_tile * 16 + ni;
  const _Float16* src = featA + ((size_t)(b * C_) + cc * 32 + cbase) * P_ + n;
  v8h v;
#pragma unroll
  for (int i = 0; i < 8; ++i) v[i] = src[(size_t)i * P_];
  *reinterpret_cast<v8h*>(featB + blk * 512 + (size_t)w * 8) = v;
}

// ---------------- K2: finalize prototypes, u_init, u, contrast loss -----------
__global__ void k_proto_final(const float* __restrict__ ws_proto, const float* __restrict__ ws_counts,
                              const float* __restrict__ protoDS, float* __restrict__ d_out,
                              _Float16* __restrict__ uT, float* __restrict__ ws_contrast) {
  __shared__ float uinit[C_ * K_];  // [c][k]
  __shared__ float colinv[K_], m[K_], uinv2[K_], part[256];
  const int tid = threadIdx.x;  // 256
  if (tid < K_) {
    float cnt = ws_counts[tid];
    d_out[1 + C_ * K_ + tid] = cnt;           // counts_batch output
    m[tid] = (cnt == 0.f) ? 0.f : 1.f;
    float s = 0.f, ic = 1.f / (cnt + 1e-6f);
    for (int c = 0; c < C_; ++c) { float v = ws_proto[tid * C_ + c] * ic; s += v * v; }
    colinv[tid] = ic / fmaxf(sqrtf(s), 1e-12f);
  }
  __syncthreads();
  {
    int c = tid;
    for (int k = 0; k < K_; ++k) {
      float ps = ws_proto[k * C_ + c];
      d_out[1 + c * K_ + k] = ps;             // prototypes_sum (C,K) output
      uinit[c * K_ + k] = ps * colinv[k];
    }
  }
  __syncthreads();
  if (tid < K_) {
    float s = 0.f;
    for (int c = 0; c < C_; ++c) {
      float v = (m[tid] == 0.f) ? protoDS[c * K_ + tid] : uinit[c * K_ + tid];
      s += v * v;
    }
    uinv2[tid] = 1.f / fmaxf(sqrtf(s), 1e-12f);
  }
  __syncthreads();
  {
    int c = tid;
    for (int k = 0; k < KP; ++k) {
      float v = 0.f;
      if (k < K_) {
        float x = (m[k] == 0.f) ? protoDS[c * K_ + k] : uinit[c * K_ + k];
        v = x * uinv2[k];
      }
      uT[k * C_ + c] = (_Float16)v;           // u^T f16, rows >=19 zero
    }
  }
  float acc = 0.f;
  for (int ij = tid; ij < K_ * K_; ij += 256) {
    int i = ij / K_, j = ij % K_;
    if (m[i] != 0.f && m[j] != 0.f) {
      float g = 0.f;
      for (int c = 0; c < C_; ++c) g += uinit[c * K_ + i] * uinit[c * K_ + j];
      acc += g;
    }
  }
  part[tid] = acc;
  __syncthreads();
  for (int s = 128; s > 0; s >>= 1) { if (tid < s) part[tid] += part[tid + s]; __syncthreads(); }
  if (tid == 0) {
    float Kn = 0.f; for (int k = 0; k < K_; ++k) Kn += m[k];
    ws_contrast[0] = (part[0] - Kn) / fmaxf(Kn * (Kn - 1.f), 1e-12f);
  }
}

// ---------------- Kd: dis = u^T featN fused with EM softmax -> p --------------
// grid: B_*(P_/128) blocks, 128 threads (4 waves, 32 n each; 4 WMMA per k-step)
__global__ void k_dis_p(const _Float16* __restrict__ featB, const _Float16* __restrict__ uT,
                        const float* __restrict__ a_f, int use_a,
                        _Float16* __restrict__ p_h, float* __restrict__ dis_f) {
  __shared__ float lds[4][KP][32];
  const int wave = threadIdx.x >> 5;
  const int lane = threadIdx.x & 31;
  const int nchunks = P_ / 128;
  const int b   = blockIdx.x / nchunks;
  const int ppb = (blockIdx.x % nchunks) * 128;
  const int nt0 = (ppb >> 4) + wave * 2;                 // first 16-wide n-tile
  const _Float16* base = featB + (size_t)b * 8 * BLKSTRIDE;
  v8f c00 = {}, c01 = {}, c10 = {}, c11 = {};
  for (int cc8 = 0; cc8 < 8; ++cc8) {
    const _Float16* blk = base + (size_t)cc8 * BLKSTRIDE + (size_t)nt0 * 512;
    v16h a0 = frag_a_rm(uT + 0 * C_ + cc8 * 32, C_);     // classes 0..15
    v16h a1 = frag_a_rm(uT + 16 * C_ + cc8 * 32, C_);    // classes 16..31
    v16h b0 = frag_b_sw(blk);                            // n-tile nt0
    v16h b1 = frag_b_sw(blk + 512);                      // n-tile nt0+1
    if (cc8 < 7) {  // WGP-scope prefetch of next k-chunk (data is L2-resident)
      __builtin_prefetch(blk + BLKSTRIDE, 0, 3);
      __builtin_prefetch(blk + BLKSTRIDE + 512, 0, 3);
    }
    c00 = __builtin_amdgcn_wmma_f32_16x16x32_f16(false, a0, false, b0, (short)0, c00, false, false);
    c01 = __builtin_amdgcn_wmma_f32_16x16x32_f16(false, a1, false, b0, (short)0, c01, false, false);
    c10 = __builtin_amdgcn_wmma_f32_16x16x32_f16(false, a0, false, b1, (short)0, c10, false, false);
    c11 = __builtin_amdgcn_wmma_f32_16x16x32_f16(false, a1, false, b1, (short)0, c11, false, false);
  }
  const int ncol = lane & 15;
  const int krow = (lane >> 4) << 3;
#pragma unroll
  for (int i = 0; i < 8; ++i) {
    lds[wave][krow + i][ncol]           = c00[i];
    lds[wave][16 + krow + i][ncol]      = c01[i];
    lds[wave][krow + i][16 + ncol]      = c10[i];
    lds[wave][16 + krow + i][16 + ncol] = c11[i];
  }
  __syncthreads();
  {
    const int nb = threadIdx.x;            // 0..127, one n per thread
    const int w = nb >> 5, ni = nb & 31;
    const int pp = ppb + nb;
    const size_t n = (size_t)b * P_ + pp;
    float e[K_]; float s = 0.f;
#pragma unroll
    for (int k = 0; k < K_; ++k) {
      float d = lds[w][k][ni];
      dis_f[n * KP + k] = d;
      float av = use_a ? a_f[pp * KP + k] : (1.f / (float)K_);
      float ev = av * __expf(KA_ * d);
      e[k] = ev; s += ev;
    }
    float inv = 1.f / s;
    _Float16* pr = p_h + n * KP;
#pragma unroll
    for (int k = 0; k < K_; ++k) pr[k] = (_Float16)(e[k] * inv);
#pragma unroll
    for (int k = K_; k < KP; ++k) pr[k] = (_Float16)0.f;  // zero pad for GEMM
  }
}

// ---------------- Ka: a = mean_b p ; also zero r accumulator ------------------
__global__ void k_a_update(const _Float16* __restrict__ p_h, float* __restrict__ a_f,
                           float* __restrict__ r) {
  const int idx = blockIdx.x * blockDim.x + threadIdx.x;
  if (idx < C_ * KP) r[idx] = 0.f;
  if (idx >= P_ * K_) return;
  const int pp = idx / K_, k = idx % K_;
  float s = 0.f;
#pragma unroll
  for (int b = 0; b < B_; ++b) s += (float)p_h[((size_t)b * P_ + pp) * KP + k];
  a_f[pp * KP + k] = s * (1.f / (float)B_);
}

// ---------------- Kr: r = featN @ p  (C x 32), partial sums + atomics ---------
// grid: 16 row-tiles x 64 n-chunks = 1024 blocks, 32 threads (one wave)
__global__ void k_r_gemm(const _Float16* __restrict__ featA, const _Float16* __restrict__ p_h,
                         float* __restrict__ r) {
  const int rt  = blockIdx.x & 15;
  const int nc  = blockIdx.x >> 4;          // 0..63, 8 chunks per batch
  const int b   = nc >> 3;
  const int pp0 = (nc & 7) * (P_ / 8);      // 2048-wide chunk, inside one batch
  const int c0  = rt * 16;
  const _Float16* fb = featA + ((size_t)(b * C_) + c0) * P_;
  v8f acc0 = {}; v8f acc1 = {};
  for (int t = 0; t < P_ / 8; t += 32) {
    const size_t n = (size_t)b * P_ + pp0 + t;
    v16h a  = frag_a_rm(fb + pp0 + t, P_);
    v16h b0 = frag_b_rm(p_h + n * KP + 0, KP);
    v16h b1 = frag_b_rm(p_h + n * KP + 16, KP);
    __builtin_prefetch(p_h + (n + 32) * KP, 0, 3);   // WGP-scope prefetch
    acc0 = __builtin_amdgcn_wmma_f32_16x16x32_f16(false, a, false, b0, (short)0, acc0, false, false);
    acc1 = __builtin_amdgcn_wmma_f32_16x16x32_f16(false, a, false, b1, (short)0, acc1, false, false);
  }
  const int lane = threadIdx.x & 31;
  const int col  = lane & 15;
  const int mrow = (lane >> 4) << 3;
#pragma unroll
  for (int i = 0; i < 8; ++i) {
    atomicAdd(&r[(c0 + mrow + i) * KP + col],      acc0[i]);
    atomicAdd(&r[(c0 + mrow + i) * KP + 16 + col], acc1[i]);
  }
}

// ---------------- Kn: column-normalize r -> uT f16 ----------------------------
__global__ void k_u_norm(const float* __restrict__ r, _Float16* __restrict__ uT) {
  __shared__ float inv[K_];
  const int tid = threadIdx.x;  // 256
  if (tid < K_) {
    float s = 0.f;
    for (int c = 0; c < C_; ++c) { float v = r[c * KP + tid]; s += v * v; }
    inv[tid] = 1.f / fmaxf(sqrtf(s), 1e-12f);
  }
  __syncthreads();
  const int c = tid;
  for (int k = 0; k < K_; ++k) uT[k * C_ + c] = (_Float16)(r[c * KP + k] * inv[k]);
}

// ---------------- Kloss: vmf + consist + clamped CE, fused over logit ---------
__global__ void k_loss(const float* __restrict__ logit, const int* __restrict__ target,
                       const _Float16* __restrict__ p_h, const float* __restrict__ dis_f,
                       const float* __restrict__ a_f, float* __restrict__ accum) {
  const size_t n = (size_t)blockIdx.x * blockDim.x + threadIdx.x;  // < N_
  const int b = (int)(n / P_), pp = (int)(n % P_);
  float vmf = 0.f, cons = 0.f, cel = 0.f, val = 0.f;
  float l[K_]; float mx = -1e30f;
#pragma unroll
  for (int k = 0; k < K_; ++k) {
    l[k] = logit[((size_t)(b * K_ + k)) * P_ + pp];
    mx = fmaxf(mx, l[k]);
  }
  float se = 0.f;
#pragma unroll
  for (int k = 0; k < K_; ++k) se += __expf(l[k] - mx);
  const float lse = mx + __logf(se);
#pragma unroll
  for (int k = 0; k < K_; ++k) {
    float pv = (float)p_h[n * KP + k];
    float d  = dis_f[n * KP + k];
    float av = a_f[pp * KP + k];
    vmf  += -pv * (__logf(av + 1e-6f) + KA_ * d);
    cons += -pv * (l[k] - lse);
  }
  {
    int t = target[n];
    bool v = (t != 255);
    int st = v ? t : 0;
    float ce = -(l[st] - lse);
    if (v) { cel = fmaxf(ce, CETH); val = 1.f; }
  }
  __shared__ float red[4][256];
  const int tid = threadIdx.x;
  red[0][tid] = vmf; red[1][tid] = cons; red[2][tid] = cel; red[3][tid] = val;
  __syncthreads();
  for (int s = 128; s > 0; s >>= 1) {
    if (tid < s) {
      red[0][tid] += red[0][tid + s]; red[1][tid] += red[1][tid + s];
      red[2][tid] += red[2][tid + s]; red[3][tid] += red[3][tid + s];
    }
    __syncthreads();
  }
  if (tid == 0) {
    atomicAdd(&accum[0], red[0][0]); atomicAdd(&accum[1], red[1][0]);
    atomicAdd(&accum[2], red[2][0]); atomicAdd(&accum[3], red[3][0]);
  }
}

__global__ void k_final(const float* __restrict__ accum, const float* __restrict__ ws_contrast,
                        float* __restrict__ d_out) {
  const float vmf = accum[0], cons = accum[1], ce = accum[2], val = accum[3];
  float loss = ce / (val + 1e-6f)
             + vmf / ((float)B_ * P_ * K_) / logf((float)K_)
             + ws_contrast[0]
             + cons / ((float)B_ * K_ * P_);
  d_out[0] = loss;
}

// ---------------- host side ---------------------------------------------------
extern "C" void kernel_launch(void* const* d_in, const int* in_sizes, int n_in,
                              void* d_out, int out_size, void* d_ws, size_t ws_size,
                              hipStream_t stream) {
  const float* logit   = (const float*)d_in[0];
  const int*   target  = (const int*)d_in[1];
  const float* feat    = (const float*)d_in[2];
  // d_in[3] = epoch (unused by the reference computation)
  const float* protoDS = (const float*)d_in[4];
  float* out = (float*)d_out;

  char* ws = (char*)d_ws;
  _Float16* featA   = (_Float16*)ws;                 ws += (size_t)B_ * C_ * P_ * 2;   // 67 MB row-major
  _Float16* featB   = (_Float16*)ws;                 ws += (size_t)B_ * C_ * P_ * 2;   // 67 MB frag-swizzled
  _Float16* p_h     = (_Float16*)ws;                 ws += (size_t)N_ * KP * 2;        // 8.4 MB
  float*    dis_f   = (float*)ws;                    ws += (size_t)N_ * KP * 4;        // 16.8 MB
  float*    a_f     = (float*)ws;                    ws += (size_t)P_ * KP * 4;        // 2 MB
  float*    r       = (float*)ws;                    ws += (size_t)C_ * KP * 4;
  _Float16* uT      = (_Float16*)ws;                 ws += (size_t)KP * C_ * 2;
  float*    ws_proto= (float*)ws;                    ws += 20 * C_ * 4;
  float*    ws_cnt  = (float*)ws;                    ws += 32 * 4;
  float*    accum   = (float*)ws;                    ws += 4 * 4;
  float*    ws_ctr  = (float*)ws;                    ws += 4;

  hipMemsetAsync(ws_proto, 0, 20 * C_ * sizeof(float), stream);
  hipMemsetAsync(ws_cnt,   0, 32 * sizeof(float),      stream);
  hipMemsetAsync(accum,    0, 4 * sizeof(float),       stream);

  // 1. segment-sum prototypes + normalize feat -> f16 (single HBM pass)
  k_proto_norm<<<B_ * (P_ / 1024), 256, 0, stream>>>(feat, target, ws_proto, ws_cnt, featA);
  // 1b. swizzle into WMMA B-fragment layout (reads L2-hot featA)
  k_swizzle<<<(int)(((size_t)B_ * C_ * P_ / 8) / 256), 256, 0, stream>>>(featA, featB);
  // 2. finalize prototypes / u / contrast; writes d_out[1..] outputs
  k_proto_final<<<1, 256, 0, stream>>>(ws_proto, ws_cnt, protoDS, out, uT, ws_ctr);
  // 3. initial E-step with a = 1/K
  k_dis_p<<<B_ * (P_ / 128), 128, 0, stream>>>(featB, uT, a_f, 0, p_h, dis_f);
  // 4. EM iterations (all GEMM traffic is L2-resident)
  for (int it = 0; it < ITERS_; ++it) {
    k_a_update<<<(P_ * K_ + 255) / 256, 256, 0, stream>>>(p_h, a_f, r);
    k_r_gemm<<<16 * 64, 32, 0, stream>>>(featA, p_h, r);
    k_u_norm<<<1, 256, 0, stream>>>(r, uT);
    k_dis_p<<<B_ * (P_ / 128), 128, 0, stream>>>(featB, uT, a_f, 1, p_h, dis_f);
  }
  // 5. fused losses over logit/p/dis/a
  k_loss<<<N_ / 256, 256, 0, stream>>>(logit, target, p_h, dis_f, a_f, accum);
  k_final<<<1, 1, 0, stream>>>(accum, ws_ctr, out);
  (void)in_sizes; (void)n_in; (void)out_size; (void)ws_size;
}
